// MemoryEfficientAttention_30236569764126
// MI455X (gfx1250) — compile-verified, matrix operand order checked by reading, not run on hardware
//
#include <hip/hip_runtime.h>

typedef __attribute__((ext_vector_type(16))) __bf16 v16bf;
typedef __attribute__((ext_vector_type(8)))  float  v8f;

#define E_DIM 1024
#define NHEAD 16
#define DHEAD 64
#define BATCH 2
#define SEQ   2048

// ---- helpers ---------------------------------------------------------------

__device__ __forceinline__ unsigned short f2bf(float x) {
  return __builtin_bit_cast(unsigned short, (__bf16)x);   // native v_cvt
}

__device__ __forceinline__ unsigned int pack2bf(float lo, float hi) {
  return (unsigned int)f2bf(lo) | ((unsigned int)f2bf(hi) << 16);
}

// Async Global->LDS 16B copy (CDNA5 VGLOBAL async op, tracked on ASYNCcnt).
// ldsaddr = LDS byte address (low 32 bits of the flat shared-pointer address;
// ISA: LDS aperture uses addr[31:0]).
__device__ __forceinline__ void async_copy_b128(void* lds, const void* gptr) {
  unsigned int l = (unsigned int)(size_t)lds;
  asm volatile("global_load_async_to_lds_b128 %0, %1, off"
               :: "v"(l), "v"(gptr) : "memory");
}

__device__ __forceinline__ void wait_async() {
  asm volatile("s_wait_asynccnt 0x0" ::: "memory");
}

// 16x32 bf16 WMMA fragment from an LDS tile stored [row][k] (ushort).
// Lane (row = lane&15, half = lane>>4) holds two contiguous 16B runs:
// K = 8*half .. +7 and K = 8*half+16 .. +23  ->  2x ds_load_b128.
// Requires: stride % 8 == 0 (elements), base 16B-aligned, kbase % 8 == 0.
__device__ __forceinline__ v16bf lds_frag(const unsigned short* base, int stride,
                                          int row, int kbase, int half) {
  union { v16bf v; uint4 q[2]; } f;
  const unsigned short* p = base + row * stride + kbase + (half << 3);
  f.q[0] = *(const uint4*)(p);
  f.q[1] = *(const uint4*)(p + 16);
  return f.v;
}

__device__ __forceinline__ v8f wmma_bf16(v16bf a, v16bf b, v8f c) {
  return __builtin_amdgcn_wmma_f32_16x16x32_bf16(false, a, false, b,
                                                 (short)0, c, false, false);
}

// ---- tiled GEMM: Out[M,N] = A[M,K] @ W[K,N] + bias -------------------------
// Block = 128 threads (4 waves), 128x64 output tile, K staged 64 deep:
// each wave owns a 32x64 sub-tile -> 16 WMMAs per barrier pair.

template <bool A_BF16, bool OUT_BF16>
__global__ __launch_bounds__(128)
void gemm128(const void* __restrict__ Ain, const float* __restrict__ W,
             const float* __restrict__ bias, void* __restrict__ Out,
             int M, int N, int K) {
  __shared__ __align__(16) unsigned short As[128][72];   // [m][k], k = 64 + pad
  __shared__ __align__(16) unsigned short Bs[64][72];    // [n][k] (transposed)
  const int tid    = threadIdx.x;
  const int wv     = tid >> 5;
  const int lane   = tid & 31;
  const int lanelo = lane & 15;
  const int half   = lane >> 4;
  const int m0 = blockIdx.y * 128;
  const int n0 = blockIdx.x * 64;

  v8f accA[4] = {};   // rows wv*32 + 0..15
  v8f accB[4] = {};   // rows wv*32 + 16..31

  for (int k0 = 0; k0 < K; k0 += 64) {
    // ---- stage A tile 128x64 ----
    if (A_BF16) {
      // straight bf16 copy: async DMA, 16B per lane per issue (1024 chunks)
#pragma unroll
      for (int i = 0; i < 8; ++i) {
        int f = i * 128 + tid;
        int r = f >> 3, ch = (f & 7) << 3;
        async_copy_b128(&As[r][ch],
                        (const unsigned short*)Ain + (size_t)(m0 + r) * K + k0 + ch);
      }
    } else {
      // f32 -> bf16: float4 load, pack, uint2 LDS store (2048 quads)
#pragma unroll
      for (int i = 0; i < 16; ++i) {
        int p = i * 128 + tid;
        int r = p >> 4, c4 = (p & 15) << 2;
        float4 xv = *(const float4*)((const float*)Ain +
                                     (size_t)(m0 + r) * K + k0 + c4);
        uint2 pk;
        pk.x = pack2bf(xv.x, xv.y);
        pk.y = pack2bf(xv.z, xv.w);
        *(uint2*)&As[r][c4] = pk;
      }
    }
    // ---- stage W tile 64x64 transposed to [n][k]: pack pairs along k ----
#pragma unroll
    for (int i = 0; i < 16; ++i) {
      int p = i * 128 + tid;
      int n = p & 63, kk = (p >> 6) << 1;
      float lo = W[(size_t)(k0 + kk) * N + n0 + n];
      float hi = W[(size_t)(k0 + kk + 1) * N + n0 + n];
      *(unsigned int*)&Bs[n][kk] = pack2bf(lo, hi);
    }
    if (A_BF16) wait_async();
    __syncthreads();

    v16bf a00 = lds_frag(&As[0][0], 72, wv * 32 + lanelo,      0,  half);
    v16bf a01 = lds_frag(&As[0][0], 72, wv * 32 + lanelo,      32, half);
    v16bf a10 = lds_frag(&As[0][0], 72, wv * 32 + 16 + lanelo, 0,  half);
    v16bf a11 = lds_frag(&As[0][0], 72, wv * 32 + 16 + lanelo, 32, half);
#pragma unroll
    for (int nt = 0; nt < 4; ++nt) {
      v16bf b0 = lds_frag(&Bs[0][0], 72, nt * 16 + lanelo, 0, half);
      accA[nt] = wmma_bf16(a00, b0, accA[nt]);
      accB[nt] = wmma_bf16(a10, b0, accB[nt]);
      v16bf b1 = lds_frag(&Bs[0][0], 72, nt * 16 + lanelo, 32, half);
      accA[nt] = wmma_bf16(a01, b1, accA[nt]);
      accB[nt] = wmma_bf16(a11, b1, accB[nt]);
    }
    __syncthreads();
  }

  // epilogue: bias add + store (C/D layout: row = r + 8*half, col = lanelo)
#pragma unroll
  for (int nt = 0; nt < 4; ++nt) {
    int col = n0 + nt * 16 + lanelo;
    float bvl = bias[col];
#pragma unroll
    for (int r = 0; r < 8; ++r) {
      int row0 = m0 + wv * 32 + r + (half << 3);
      int row1 = row0 + 16;
      float v0 = accA[nt][r] + bvl;
      float v1 = accB[nt][r] + bvl;
      if (OUT_BF16) {
        ((unsigned short*)Out)[(size_t)row0 * N + col] = f2bf(v0);
        ((unsigned short*)Out)[(size_t)row1 * N + col] = f2bf(v1);
      } else {
        ((float*)Out)[(size_t)row0 * N + col] = v0;
        ((float*)Out)[(size_t)row1 * N + col] = v1;
      }
    }
  }
}

// ---- flash attention over one (batch, head, 64-query tile) -----------------

__global__ __launch_bounds__(128)
void attn64(const unsigned short* __restrict__ Q,
            const unsigned short* __restrict__ K,
            const unsigned short* __restrict__ V,
            unsigned short* __restrict__ Ctx) {
  __shared__ __align__(16) unsigned short Qs[64][72];      // [q][d]
  __shared__ __align__(16) unsigned short Ks[64][72];      // [key][d]
  __shared__ __align__(16) unsigned short Vs[64][72];      // [d][key] (transposed)
  __shared__ __align__(16) unsigned short Ps[4][16][72];   // per-wave P [q][key]

  const int tid    = threadIdx.x;
  const int wv     = tid >> 5;
  const int lane   = tid & 31;
  const int lanelo = lane & 15;
  const int half   = lane >> 4;

  const int q0 = blockIdx.x * 64;
  const int b  = blockIdx.y / NHEAD;
  const int h  = blockIdx.y % NHEAD;
  const size_t rowbase = (size_t)b * SEQ;
  const int cbase = h * DHEAD;

  // stage Q tile via async DMA (64 rows x 128B = 512 x 16B chunks)
#pragma unroll
  for (int i = 0; i < 4; ++i) {
    int f = i * 128 + tid;
    int r = f >> 3, ch = (f & 7) << 3;
    async_copy_b128(&Qs[r][ch], &Q[(rowbase + q0 + r) * E_DIM + cbase + ch]);
  }
  wait_async();
  __syncthreads();
  v16bf aq0 = lds_frag(&Qs[0][0], 72, wv * 16 + lanelo, 0,  half);
  v16bf aq1 = lds_frag(&Qs[0][0], 72, wv * 16 + lanelo, 32, half);

  float mrow[8], lrow[8];
#pragma unroll
  for (int r = 0; r < 8; ++r) { mrow[r] = -3.0e38f; lrow[r] = 0.f; }
  v8f o[4] = {};

  const float scale = 0.125f;   // 1/sqrt(DH)

  for (int kt = 0; kt < SEQ / 64; ++kt) {
    const int kr0 = kt * 64;
    // stage K [key][d] via async DMA (straight copy)
#pragma unroll
    for (int i = 0; i < 4; ++i) {
      int f = i * 128 + tid;
      int r = f >> 3, ch = (f & 7) << 3;
      async_copy_b128(&Ks[r][ch], &K[(rowbase + kr0 + r) * E_DIM + cbase + ch]);
    }
    // stage V transposed [d][key]: pack pairs along key (coalesced over d)
#pragma unroll
    for (int i = 0; i < 16; ++i) {
      int p = i * 128 + tid;
      int d = p & 63, kk = (p >> 6) << 1;
      unsigned int lo = V[(rowbase + kr0 + kk)     * E_DIM + cbase + d];
      unsigned int hi = V[(rowbase + kr0 + kk + 1) * E_DIM + cbase + d];
      *(unsigned int*)&Vs[d][kk] = lo | (hi << 16);
    }
    if (kt + 1 < SEQ / 64) {   // warm next V tile (global_prefetch_b8)
      __builtin_prefetch(&V[(rowbase + kr0 + 64 + (tid & 63)) * E_DIM + cbase], 0, 0);
    }
    wait_async();
    __syncthreads();

    // scores: 16 q-rows x 64 keys per wave (8 WMMAs)
    v8f sc[4] = {};
#pragma unroll
    for (int nt = 0; nt < 4; ++nt) {
      v16bf b0 = lds_frag(&Ks[0][0], 72, nt * 16 + lanelo, 0,  half);
      sc[nt] = wmma_bf16(aq0, b0, sc[nt]);
      v16bf b1 = lds_frag(&Ks[0][0], 72, nt * 16 + lanelo, 32, half);
      sc[nt] = wmma_bf16(aq1, b1, sc[nt]);
    }

    // online softmax; each lane owns 8 rows (r + 8*half), col group = lanelo
#pragma unroll
    for (int r = 0; r < 8; ++r) {
      float s0 = sc[0][r] * scale, s1 = sc[1][r] * scale;
      float s2 = sc[2][r] * scale, s3 = sc[3][r] * scale;
      float mx = fmaxf(fmaxf(s0, s1), fmaxf(s2, s3));
#pragma unroll
      for (int msk = 1; msk < 16; msk <<= 1) mx = fmaxf(mx, __shfl_xor(mx, msk));
      float mnew = fmaxf(mrow[r], mx);
      float corr = __expf(mrow[r] - mnew);
      float p0 = __expf(s0 - mnew), p1 = __expf(s1 - mnew);
      float p2 = __expf(s2 - mnew), p3 = __expf(s3 - mnew);
      float sum = p0 + p1 + p2 + p3;
#pragma unroll
      for (int msk = 1; msk < 16; msk <<= 1) sum += __shfl_xor(sum, msk);
      lrow[r] = lrow[r] * corr + sum;
      mrow[r] = mnew;
      o[0][r] *= corr; o[1][r] *= corr; o[2][r] *= corr; o[3][r] *= corr;
      int prow = r + (half << 3);
      Ps[wv][prow][lanelo +  0] = f2bf(p0);
      Ps[wv][prow][lanelo + 16] = f2bf(p1);
      Ps[wv][prow][lanelo + 32] = f2bf(p2);
      Ps[wv][prow][lanelo + 48] = f2bf(p3);
    }
    __syncthreads();   // P visible; also fences Ks before next staging

    // O += P @ V  (K-dim = 64 keys -> 2 k-steps, 4 d-tiles: 8 WMMAs)
    v16bf ap0 = lds_frag(&Ps[wv][0][0], 72, lanelo, 0,  half);
    v16bf ap1 = lds_frag(&Ps[wv][0][0], 72, lanelo, 32, half);
#pragma unroll
    for (int nt = 0; nt < 4; ++nt) {
      v16bf b0 = lds_frag(&Vs[0][0], 72, nt * 16 + lanelo, 0,  half);
      o[nt] = wmma_bf16(ap0, b0, o[nt]);
      v16bf b1 = lds_frag(&Vs[0][0], 72, nt * 16 + lanelo, 32, half);
      o[nt] = wmma_bf16(ap1, b1, o[nt]);
    }
    __syncthreads();
  }

  // normalize and write context (bf16, [B,S,H*DH] layout)
#pragma unroll
  for (int r = 0; r < 8; ++r) {
    float inv = 1.0f / lrow[r];
    int row = q0 + wv * 16 + r + (half << 3);
#pragma unroll
    for (int nt = 0; nt < 4; ++nt) {
      Ctx[(rowbase + row) * E_DIM + cbase + nt * 16 + lanelo] = f2bf(o[nt][r] * inv);
    }
  }
}

// ---- launch ----------------------------------------------------------------

extern "C" void kernel_launch(void* const* d_in, const int* in_sizes, int n_in,
                              void* d_out, int out_size, void* d_ws, size_t ws_size,
                              hipStream_t stream) {
  const float* q  = (const float*)d_in[0];
  const float* k  = (const float*)d_in[1];
  const float* v  = (const float*)d_in[2];
  const float* Wq = (const float*)d_in[3];
  const float* bq = (const float*)d_in[4];
  const float* Wk = (const float*)d_in[5];
  const float* bk = (const float*)d_in[6];
  const float* Wv = (const float*)d_in[7];
  const float* bv = (const float*)d_in[8];
  const float* Wo = (const float*)d_in[9];
  const float* bo = (const float*)d_in[10];

  const int M = BATCH * SEQ;                       // 4096 rows
  unsigned short* Qb = (unsigned short*)d_ws;      // bf16 Q   (8 MB)
  unsigned short* Kb = Qb + (size_t)M * E_DIM;     // bf16 K   (8 MB)
  unsigned short* Vb = Kb + (size_t)M * E_DIM;     // bf16 V   (8 MB)
  unsigned short* Cb = Vb + (size_t)M * E_DIM;     // bf16 ctx (8 MB)

  dim3 g(E_DIM / 64, M / 128);                     // (16, 32)
  gemm128<false, true><<<g, 128, 0, stream>>>(q, Wq, bq, Qb, M, E_DIM, E_DIM);
  gemm128<false, true><<<g, 128, 0, stream>>>(k, Wk, bk, Kb, M, E_DIM, E_DIM);
  gemm128<false, true><<<g, 128, 0, stream>>>(v, Wv, bv, Vb, M, E_DIM, E_DIM);

  dim3 ga(SEQ / 64, BATCH * NHEAD);                // (32, 32)
  attn64<<<ga, 128, 0, stream>>>(Qb, Kb, Vb, Cb);

  gemm128<true, false><<<g, 128, 0, stream>>>(Cb, Wo, bo, (float*)d_out,
                                              M, E_DIM, E_DIM);
}
